// SuperSimpleAttention_7679401525936
// MI455X (gfx1250) — matrix operand order repaired and checked by reading, not compiled, not run on hardware
//
#include <hip/hip_runtime.h>
#include <stdint.h>

typedef __attribute__((ext_vector_type(16))) __bf16 v16bf;
typedef __attribute__((ext_vector_type(8)))  float  v8f;
typedef int v4i_vs __attribute__((vector_size(16)));   // matches builtin param type

#if defined(__AMDGCN__) && __has_builtin(__builtin_amdgcn_global_load_async_to_lds_b128) && __has_builtin(__builtin_amdgcn_s_wait_asynccnt)
#define USE_ASYNC 1
#else
#define USE_ASYNC 0
#endif

static __device__ __forceinline__ uint16_t f32_to_bf16(float f) {
  uint32_t u = __float_as_uint(f);
  uint32_t r = u + 0x7FFFu + ((u >> 16) & 1u);   // round-to-nearest-even
  return (uint16_t)(r >> 16);
}
static __device__ __forceinline__ float bf16_to_f32(uint16_t h) {
  return __uint_as_float(((uint32_t)h) << 16);
}

// 128-bit global -> LDS copy: async DMA (ASYNCcnt) when available, else sync.
static __device__ __forceinline__ void cp_b128(uint16_t* l, const uint16_t* g) {
#if USE_ASYNC
  __builtin_amdgcn_global_load_async_to_lds_b128(
      (__attribute__((address_space(1))) v4i_vs*)g,
      (__attribute__((address_space(3))) v4i_vs*)l, 0, 0);
#else
  *(uint4*)l = *(const uint4*)g;
#endif
}
static __device__ __forceinline__ void wait_cp() {
#if USE_ASYNC
  __builtin_amdgcn_s_wait_asynccnt(0);
#endif
}

union Frag {
  v16bf    v;
  uint4    q[2];
  uint16_t u[16];
};

// ---------------- f32 -> bf16 conversion (vectorized x4) ----------------
__global__ __launch_bounds__(256)
void cvt_f32_bf16(const float* __restrict__ src, uint16_t* __restrict__ dst, long n4) {
  long i = (long)blockIdx.x * blockDim.x + threadIdx.x;
  if (i >= n4) return;
  float4 f = ((const float4*)src)[i];
  union { uint16_t h[4]; uint2 q; } t;
  t.h[0] = f32_to_bf16(f.x);
  t.h[1] = f32_to_bf16(f.y);
  t.h[2] = f32_to_bf16(f.z);
  t.h[3] = f32_to_bf16(f.w);
  ((uint2*)dst)[i] = t.q;
}

// ---------------- generic bf16 WMMA GEMM with LDS double buffering ------
// Out[m,n] = scale * sum_k A[m,k] * Bm[n,k]
// Block: 128x128 tile, 4 waves (2x2), each wave 64x64 = 4x4 WMMA tiles
// (16 wmma per 16 ds_load_b128 -> 1:1 math:LDS ratio). K consumed in
// 64-wide slabs staged in LDS (double buffered, async global->LDS DMA
// tracked by ASYNCcnt).
// MASK:   causal mask at store (col>row => -1), bf16 out
// TRANS:  store transposed bf16 (Out[n*ldo + m]) with packed 16B stores
// F32OUT: f32 output (+ optional BIAS[col])
template<int MASK, int TRANS, int F32OUT, int BIAS>
__global__ __launch_bounds__(128)
void wmma_gemm(const uint16_t* __restrict__ A, int lda, long long sA,
               const uint16_t* __restrict__ Bm, int ldb, long long sB,
               void* __restrict__ Out, int ldo, long long sO,
               const float* __restrict__ bias, float scale, int K)
{
  constexpr int LDT = 72;                 // padded LDS row stride (elements)
  __shared__ __align__(16) uint16_t lA[2][128 * LDT];
  __shared__ __align__(16) uint16_t lB[2][128 * LDT];

  const int z = blockIdx.z;
  A  += (long long)z * sA;
  Bm += (long long)z * sB;

  const int lane  = threadIdx.x & 31;
  const int w     = threadIdx.x >> 5;              // 0..3
  const int wmRow = (w >> 1) * 64;                 // wave M origin in tile
  const int wnCol = (w & 1) * 64;                  // wave N origin in tile
  const int bm0   = blockIdx.y * 128;
  const int bn0   = blockIdx.x * 128;
  const int l15   = lane & 15;
  const int lh    = lane >> 4;                     // lane half (0/1)

  // stage a 64-deep K slab of the A (128 rows) and B (128 cols) tiles
  auto copy_tiles = [&](int buf, int kk) {
#pragma unroll
    for (int i = 0; i < 8; ++i) {
      const int flat = (int)threadIdx.x + i * 128;  // 1024 x 16B for A
      const int r = flat >> 3, s8 = (flat & 7) * 8;
      cp_b128(&lA[buf][r * LDT + s8], A + (long long)(bm0 + r) * lda + kk + s8);
    }
#pragma unroll
    for (int i = 0; i < 8; ++i) {
      const int flat = (int)threadIdx.x + i * 128;  // 1024 x 16B for B
      const int r = flat >> 3, s8 = (flat & 7) * 8;
      cp_b128(&lB[buf][r * LDT + s8], Bm + (long long)(bn0 + r) * ldb + kk + s8);
    }
  };

  v8f acc[4][4];
  v8f zero = {};
#pragma unroll
  for (int i = 0; i < 4; ++i)
#pragma unroll
    for (int j = 0; j < 4; ++j) acc[i][j] = zero;

  const int nstep = K >> 6;                        // K / 64 slabs
  copy_tiles(0, 0);
  for (int s = 0; s < nstep; ++s) {
    const int cur = s & 1;
    wait_cp();
    __syncthreads();                               // slab `cur` ready; prev compute done
    if (s + 1 < nstep) copy_tiles(cur ^ 1, (s + 1) << 6);

    const uint16_t* lac = lA[cur];
    const uint16_t* lbc = lB[cur];
#pragma unroll
    for (int c = 0; c < 2; ++c) {                  // two K=32 chunks per slab
      Frag a[4], b[4];
#pragma unroll
      for (int mt = 0; mt < 4; ++mt) {
        const uint16_t* p = lac + (wmRow + mt * 16 + l15) * LDT + c * 32 + lh * 8;
        a[mt].q[0] = *(const uint4*)p;
        a[mt].q[1] = *(const uint4*)(p + 16);
      }
#pragma unroll
      for (int nt = 0; nt < 4; ++nt) {
        const uint16_t* p = lbc + (wnCol + nt * 16 + l15) * LDT + c * 32 + lh * 8;
        b[nt].q[0] = *(const uint4*)p;
        b[nt].q[1] = *(const uint4*)(p + 16);
      }
#pragma unroll
      for (int mt = 0; mt < 4; ++mt)
#pragma unroll
        for (int nt = 0; nt < 4; ++nt)
          acc[mt][nt] = __builtin_amdgcn_wmma_f32_16x16x32_bf16(
              false, a[mt].v, false, b[nt].v, (short)0, acc[mt][nt], false, false);
    }
  }

  // store: C/D layout — lane col = l15, rows = lh*8 + i
  const int bm = bm0 + wmRow;
  const int bn = bn0 + wnCol;
#pragma unroll
  for (int mt = 0; mt < 4; ++mt) {
#pragma unroll
    for (int nt = 0; nt < 4; ++nt) {
      const int col  = bn + nt * 16 + l15;
      const int row0 = bm + mt * 16 + lh * 8;
      if (F32OUT) {
        float* O = (float*)Out + (long long)z * sO;
        const float badd = BIAS ? bias[col] : 0.0f;
#pragma unroll
        for (int i = 0; i < 8; ++i)
          O[(long long)(row0 + i) * ldo + col] = acc[mt][nt][i] * scale + badd;
      } else if (TRANS) {
        uint16_t* O = (uint16_t*)Out + (long long)z * sO;
        union { uint16_t h[8]; uint4 q; } t;
#pragma unroll
        for (int i = 0; i < 8; ++i) t.h[i] = f32_to_bf16(acc[mt][nt][i] * scale);
        *(uint4*)(O + (long long)col * ldo + row0) = t.q;   // 8 consecutive rows
      } else {
        uint16_t* O = (uint16_t*)Out + (long long)z * sO;
#pragma unroll
        for (int i = 0; i < 8; ++i) {
          float val = acc[mt][nt][i] * scale;
          if (MASK) val = (col <= row0 + i) ? val : -1.0f;
          O[(long long)(row0 + i) * ldo + col] = f32_to_bf16(val);
        }
      }
    }
  }
}

// ---------------- row normalization: att[r,:] /= (sum(att[r,:]) + 1e-5) ----
// one wave per 2048-element bf16 row; 8 rows per 256-thread block
__global__ __launch_bounds__(256)
void rownorm(uint16_t* __restrict__ att, int rowlen /*2048*/) {
  const int lane = threadIdx.x & 31;
  const int w    = threadIdx.x >> 5;
  uint16_t* p = att + ((long long)blockIdx.x * 8 + w) * rowlen;
  uint4 vbuf[8];
  float s = 0.0f;
#pragma unroll
  for (int it = 0; it < 8; ++it) {
    vbuf[it] = *(const uint4*)(p + (it * 32 + lane) * 8);
    const uint16_t* u = (const uint16_t*)&vbuf[it];
#pragma unroll
    for (int j = 0; j < 8; ++j) s += bf16_to_f32(u[j]);
  }
#pragma unroll
  for (int off = 16; off > 0; off >>= 1) s += __shfl_xor(s, off, 32);
  const float rs = 1.0f / (s + 1e-5f);
#pragma unroll
  for (int it = 0; it < 8; ++it) {
    union { uint16_t h[8]; uint4 q; } t;
    const uint16_t* u = (const uint16_t*)&vbuf[it];
#pragma unroll
    for (int j = 0; j < 8; ++j) t.h[j] = f32_to_bf16(bf16_to_f32(u[j]) * rs);
    *(uint4*)(p + (it * 32 + lane) * 8) = t.q;
  }
}

// ---------------- host launch ----------------
extern "C" void kernel_launch(void* const* d_in, const int* in_sizes, int n_in,
                              void* d_out, int out_size, void* d_ws, size_t ws_size,
                              hipStream_t stream) {
  (void)in_sizes; (void)n_in; (void)out_size; (void)ws_size;
  constexpr int Bb = 8, T = 2048, C = 1024;

  const float* x  = (const float*)d_in[0];
  const float* Wq = (const float*)d_in[1];
  const float* Wk = (const float*)d_in[2];
  const float* Wv = (const float*)d_in[3];
  const float* Wo = (const float*)d_in[4];
  const float* bo = (const float*)d_in[5];

  char* ws = (char*)d_ws;
  size_t off = 0;
  auto alloc = [&](size_t bytes) -> char* {
    char* p = ws + off;
    off += (bytes + 255) & ~(size_t)255;
    return p;
  };
  uint16_t* xb  = (uint16_t*)alloc((size_t)Bb * T * C * 2);   // x bf16
  uint16_t* qb  = (uint16_t*)alloc((size_t)Bb * T * C * 2);   // q (later reused as y_attn)
  uint16_t* kb  = (uint16_t*)alloc((size_t)Bb * T * C * 2);   // k
  uint16_t* vT  = (uint16_t*)alloc((size_t)Bb * C * T * 2);   // v transposed [b, d, t]
  uint16_t* att = (uint16_t*)alloc((size_t)Bb * T * T * 2);   // attention matrix
  uint16_t* Wqb = (uint16_t*)alloc((size_t)C * C * 2);
  uint16_t* Wkb = (uint16_t*)alloc((size_t)C * C * 2);
  uint16_t* Wvb = (uint16_t*)alloc((size_t)C * C * 2);
  uint16_t* Wob = (uint16_t*)alloc((size_t)C * C * 2);
  uint16_t* yb  = qb;  // q is dead after att GEMM; reuse for att@v result

  // 1) convert inputs to bf16
  {
    long n4 = (long)Bb * T * C / 4;
    cvt_f32_bf16<<<dim3((unsigned)((n4 + 255) / 256)), 256, 0, stream>>>(x, xb, n4);
    long w4 = (long)C * C / 4;
    unsigned g = (unsigned)((w4 + 255) / 256);
    cvt_f32_bf16<<<g, 256, 0, stream>>>(Wq, Wqb, w4);
    cvt_f32_bf16<<<g, 256, 0, stream>>>(Wk, Wkb, w4);
    cvt_f32_bf16<<<g, 256, 0, stream>>>(Wv, Wvb, w4);
    cvt_f32_bf16<<<g, 256, 0, stream>>>(Wo, Wob, w4);
  }

  const long long sXC = (long long)T * C;   // per-batch stride of [T,C] tensors
  const long long sTT = (long long)T * T;
  const long long sCT = (long long)C * T;
  dim3 blk(128);                            // 4 waves per block

  // 2) projections: q = 0.1*x*Wq^T ; k = 0.1*x*Wk^T ; vT = (x*Wv^T)^T
  dim3 gProj(C / 128, T / 128, Bb);
  wmma_gemm<0, 0, 0, 0><<<gProj, blk, 0, stream>>>(xb, C, sXC, Wqb, C, 0, qb, C, sXC, nullptr, 0.1f, C);
  wmma_gemm<0, 0, 0, 0><<<gProj, blk, 0, stream>>>(xb, C, sXC, Wkb, C, 0, kb, C, sXC, nullptr, 0.1f, C);
  wmma_gemm<0, 1, 0, 0><<<gProj, blk, 0, stream>>>(xb, C, sXC, Wvb, C, 0, vT, T, sCT, nullptr, 1.0f, C);

  // 3) att = q @ k^T with causal mask (-1 above diagonal)
  dim3 gAtt(T / 128, T / 128, Bb);
  wmma_gemm<1, 0, 0, 0><<<gAtt, blk, 0, stream>>>(qb, C, sXC, kb, C, sXC, att, T, sTT, nullptr, 1.0f, C);

  // 4) linear row normalization in place
  rownorm<<<dim3(Bb * T / 8), dim3(256), 0, stream>>>(att, T);

  // 5) y = att @ v  (v supplied transposed as B operand)
  wmma_gemm<0, 0, 0, 0><<<gProj, blk, 0, stream>>>(att, T, sTT, vT, T, sCT, yb, C, sXC, nullptr, 1.0f, T);

  // 6) out = y @ Wo^T + bo  (f32 output)
  wmma_gemm<0, 0, 1, 1><<<gProj, blk, 0, stream>>>(yb, C, sXC, Wob, C, 0, d_out, C, sXC, bo, 1.0f, C);
}
